// MoLECNNRewardModel_36000415875264
// MI455X (gfx1250) — matrix operand order repaired and checked by reading, not compiled, not run on hardware
//
#include <hip/hip_runtime.h>
#include <hip/hip_bf16.h>

typedef __bf16 bf16;
typedef __attribute__((ext_vector_type(16))) __bf16 v16bf;
typedef __attribute__((ext_vector_type(8)))  float  v8f;

#define LDSW 40   // LDS tile row stride in halves (32 + 8 pad)

static __device__ __forceinline__ bf16 f2bf(float f) { return (bf16)f; }

// ---- CDNA5 async global->LDS copy (ASYNCcnt-tracked, no VGPR round trip) ----
static __device__ __forceinline__ void async_cp16(unsigned int lds, unsigned long long ga) {
    asm volatile("global_load_async_to_lds_b128 %0, %1, off"
                 :: "v"(lds), "v"(ga) : "memory");
}
static __device__ __forceinline__ void wait_async(int pending) {
    if (pending) asm volatile("s_wait_asynccnt 0x4" ::: "memory");
    else         asm volatile("s_wait_asynccnt 0x0" ::: "memory");
}

// ---------------------------------------------------------------------------
// WMMA fragment math shared by both GEMM loop flavors.
// Wave tile 32(M) x 64(N) = 2x4 16x16 tiles, K step 32.
// ---------------------------------------------------------------------------
static __device__ __forceinline__ void wmma_step(
    const bf16* __restrict__ As, const bf16* __restrict__ Bs,
    int wm, int wn, int lh, int lg, v8f acc[2][4])
{
    // A frag: M = lane%16; halves 0..7 -> K = 8*lg.., halves 8..15 -> K = 16+8*lg..
    v16bf af[2];
    #pragma unroll
    for (int i = 0; i < 2; ++i) {
        const bf16* base = As + (wm * 32 + i * 16 + lh) * LDSW;
        union { int4 q[2]; v16bf v; } u;
        u.q[0] = *(const int4*)(base + 8 * lg);
        u.q[1] = *(const int4*)(base + 16 + 8 * lg);
        af[i] = u.v;
    }
    // B frag: N = lane%16; halves 0..7 -> K = 8*lg.., halves 8..15 -> K = 16+8*lg..
    v16bf bfr[4];
    #pragma unroll
    for (int j = 0; j < 4; ++j) {
        const bf16* base = Bs + (wn * 64 + j * 16 + lh) * LDSW;
        union { int4 q[2]; v16bf v; } u;
        u.q[0] = *(const int4*)(base + 8 * lg);
        u.q[1] = *(const int4*)(base + 16 + 8 * lg);
        bfr[j] = u.v;
    }
    #pragma unroll
    for (int i = 0; i < 2; ++i)
        #pragma unroll
        for (int j = 0; j < 4; ++j)
            acc[i][j] = __builtin_amdgcn_wmma_f32_16x16x32_bf16(
                false, af[i], false, bfr[j], (short)0, acc[i][j], false, false);
}

// ---------------------------------------------------------------------------
// Synchronous GEMM loop (handles partial row counts) — used by k_gemm_p only.
// C[M][N] += A[M][K] * B[N][K]^T
// ---------------------------------------------------------------------------
static __device__ __forceinline__ void gemm_acc_sync(
    bf16* __restrict__ As, bf16* __restrict__ Bs,
    const bf16* __restrict__ A, int lda, int Mrows,
    const bf16* __restrict__ B, int ldb, int Nrows,
    int K, int m0, int n0, v8f acc[2][4])
{
    const int tid = threadIdx.x;
    const int wave = tid >> 5, lane = tid & 31;
    const int wm = wave >> 1, wn = wave & 1;
    const int lh = lane & 15, lg = lane >> 4;
    const int ldRow = tid >> 1;
    const int ldCol = (tid & 1) << 4;

    for (int k0 = 0; k0 < K; k0 += 32) {
        __syncthreads();
        int4 a0 = {0,0,0,0}, a1 = {0,0,0,0};
        int gm = m0 + ldRow;
        if (gm < Mrows) {
            const int4* s = (const int4*)(A + (size_t)gm * lda + k0 + ldCol);
            a0 = s[0]; a1 = s[1];
        }
        int4 b0 = {0,0,0,0}, b1 = {0,0,0,0};
        int gn = n0 + ldRow;
        if (gn < Nrows) {
            const int4* s = (const int4*)(B + (size_t)gn * ldb + k0 + ldCol);
            b0 = s[0]; b1 = s[1];
        }
        {
            int4* d = (int4*)(As + ldRow * LDSW + ldCol);
            d[0] = a0; d[1] = a1;
            int4* e = (int4*)(Bs + ldRow * LDSW + ldCol);
            e[0] = b0; e[1] = b1;
        }
        __syncthreads();
        wmma_step(As, Bs, wm, wn, lh, lg, acc);
    }
}

// ---------------------------------------------------------------------------
// Async double-buffered GEMM loop: full 128-row tiles only (no bounds).
// Next tile's global->LDS DMA overlaps current tile's WMMAs.
// ---------------------------------------------------------------------------
static __device__ __forceinline__ void gemm_acc_async(
    bf16* __restrict__ As0, bf16* __restrict__ Bs0,
    bf16* __restrict__ As1, bf16* __restrict__ Bs1,
    const bf16* __restrict__ A, int lda,
    const bf16* __restrict__ B, int ldb,
    int K, int m0, int n0, v8f acc[2][4])
{
    const int tid = threadIdx.x;
    const int wave = tid >> 5, lane = tid & 31;
    const int wm = wave >> 1, wn = wave & 1;
    const int lh = lane & 15, lg = lane >> 4;
    const int ldRow = tid >> 1;
    const int ldCol = (tid & 1) << 4;

    unsigned long long gaA = (unsigned long long)(uintptr_t)(A + (size_t)(m0 + ldRow) * lda + ldCol);
    unsigned long long gaB = (unsigned long long)(uintptr_t)(B + (size_t)(n0 + ldRow) * ldb + ldCol);
    const unsigned int lA0 = (unsigned int)(uintptr_t)(As0 + ldRow * LDSW + ldCol);
    const unsigned int lA1 = (unsigned int)(uintptr_t)(As1 + ldRow * LDSW + ldCol);
    const unsigned int lB0 = (unsigned int)(uintptr_t)(Bs0 + ldRow * LDSW + ldCol);
    const unsigned int lB1 = (unsigned int)(uintptr_t)(Bs1 + ldRow * LDSW + ldCol);

    const int NIT = K >> 5;
    __syncthreads();                       // all waves done reading LDS (prev call)
    async_cp16(lA0, gaA); async_cp16(lA0 + 32, gaA + 32);
    async_cp16(lB0, gaB); async_cp16(lB0 + 32, gaB + 32);

    for (int it = 0; it < NIT; ++it) {
        const int cur = it & 1;
        gaA += 64; gaB += 64;              // next K chunk (32 halves)
        const bool more = (it + 1 < NIT);
        if (more) {
            const unsigned int nA = cur ? lA0 : lA1;
            const unsigned int nB = cur ? lB0 : lB1;
            async_cp16(nA, gaA); async_cp16(nA + 32, gaA + 32);
            async_cp16(nB, gaB); async_cp16(nB + 32, gaB + 32);
        }
        wait_async(more ? 4 : 0);          // own cur-tile DMAs complete
        __syncthreads();                   // => every wave's cur tile resident
        wmma_step(cur ? As1 : As0, cur ? Bs1 : Bs0, wm, wn, lh, lg, acc);
        __syncthreads();                   // all reads of cur done before overwrite
    }
}

// ---------------------------------------------------------------------------
// Gating: rw = (onehot(argmax softmax(z/TAU)) - soft) + soft
// ---------------------------------------------------------------------------
__global__ __launch_bounds__(128) void k_gating(
    const float* __restrict__ user, const float* __restrict__ gw1,
    const float* __restrict__ gb1,  const float* __restrict__ gw2,
    const float* __restrict__ gb2,  float* __restrict__ rw)
{
    int b = blockIdx.x, j = threadIdx.x;
    __shared__ float g[128];
    __shared__ float z[8];
    const float* u = user + b * 128;
    float a = gb1[j];
    for (int i = 0; i < 128; ++i) a += u[i] * gw1[i * 128 + j];
    g[j] = a > 0.f ? a : 0.f;
    __syncthreads();
    if (j < 8) {
        float zz = gb2[j];
        for (int i = 0; i < 128; ++i) zz += g[i] * gw2[i * 8 + j];
        z[j] = zz;
    }
    __syncthreads();
    if (j == 0) {
        float mx = z[0]; int am = 0;
        for (int e = 1; e < 8; ++e) if (z[e] > mx) { mx = z[e]; am = e; }
        float ex[8], den = 0.f;
        for (int e = 0; e < 8; ++e) { ex[e] = __expf((z[e] - mx) * 0.5f); den += ex[e]; }
        for (int e = 0; e < 8; ++e) {
            float soft = ex[e] / den;
            float hard = (e == am) ? 1.f : 0.f;
            rw[b * 8 + e] = (hard - soft) + soft;
        }
    }
}

// ------------------------- packing kernels ---------------------------------
__global__ void k_cvt_bf16(const float* __restrict__ in, bf16* __restrict__ out, long long n) {
    long long i = (long long)blockIdx.x * blockDim.x + threadIdx.x;
    long long stride = (long long)gridDim.x * blockDim.x;
    for (; i < n; i += stride) out[i] = f2bf(in[i]);
}

__global__ void k_pack_opwT(const float* __restrict__ w, bf16* __restrict__ out) {
    int idx = blockIdx.x * blockDim.x + threadIdx.x;   // 512*512
    int o = idx >> 9, i = idx & 511;
    out[idx] = f2bf(w[i * 512 + o]);                   // out[o][i] = w[i][o]
}

__global__ void k_pack_AallT(const float* __restrict__ As_, const float* __restrict__ Ae,
                             bf16* __restrict__ out) {
    int idx = blockIdx.x * blockDim.x + threadIdx.x;   // 96*512
    int rp = idx >> 9, i = idx & 511;
    float v = 0.f;
    if (rp < 8) v = As_[i * 8 + rp];
    else if (rp < 72) {
        int e = (rp - 8) >> 3, r = (rp - 8) & 7;
        v = Ae[((size_t)e * 512 + i) * 8 + r];
    }
    out[idx] = f2bf(v);                                // out[rp][i]
}

__global__ void k_pack_BcT(const float* __restrict__ Bs_, const float* __restrict__ Be,
                           const float* __restrict__ rw, bf16* __restrict__ out) {
    int idx = blockIdx.x * blockDim.x + threadIdx.x;   // 64*512*96
    int b = idx / (512 * 96);
    int rem = idx - b * 512 * 96;
    int o = rem / 96, rp = rem - o * 96;
    float v = 0.f;
    if (rp < 8) v = Bs_[rp * 512 + o];
    else if (rp < 72) {
        int e = (rp - 8) >> 3, r = (rp - 8) & 7;
        v = rw[b * 8 + e] * Be[(size_t)(e * 8 + r) * 512 + o];
    }
    out[idx] = f2bf(v);                                // out[b][o][rp]
}

__global__ void k_pack_convw(const float* __restrict__ cw, bf16* __restrict__ out) {
    int idx = blockIdx.x * blockDim.x + threadIdx.x;   // 3*512*512
    int k = idx / (512 * 512);
    int rem = idx - k * 512 * 512;
    int o = rem >> 9, i = rem & 511;
    out[idx] = f2bf(cw[((size_t)o * 512 + i) * 3 + k]);  // out[k][o][i]
}

__global__ void k_zero_pads(bf16* __restrict__ hpad, bf16* __restrict__ h2pad) {
    int idx = blockIdx.x * blockDim.x + threadIdx.x;   // 64*2*512
    int b = idx >> 10, r = (idx >> 9) & 1, o = idx & 511;
    size_t off = ((size_t)b * 514 + (r ? 513 : 0)) * 512 + o;
    hpad[off]  = f2bf(0.f);
    h2pad[off] = f2bf(0.f);
}

// ---------------------------------------------------------------------------
// G1: rank projection. P[b][t][rp] = sum_i obs[b][t][i]*Aall[i][rp]
// (M = rp: 96 rows, rows 72..95 zero; N = t; K = 512)
// ---------------------------------------------------------------------------
__global__ __launch_bounds__(256) void k_gemm_p(
    const bf16* __restrict__ AallT, const bf16* __restrict__ obs_bf,
    bf16* __restrict__ P)
{
    __shared__ __align__(16) bf16 As[128 * LDSW];
    __shared__ __align__(16) bf16 Bs[128 * LDSW];
    int b = blockIdx.y, n0 = blockIdx.x * 128;
    v8f acc[2][4];
    v8f z = {0,0,0,0,0,0,0,0};
    for (int i = 0; i < 2; ++i) for (int j = 0; j < 4; ++j) acc[i][j] = z;

    gemm_acc_sync(As, Bs, AallT, 512, 96,
                  obs_bf + (size_t)b * 512 * 512, 512, 512,
                  512, 0, n0, acc);

    const int wave = threadIdx.x >> 5, lane = threadIdx.x & 31;
    const int wm = wave >> 1, wn = wave & 1, lh = lane & 15, lg = lane >> 4;
    bf16* Pb = P + (size_t)b * 512 * 96;
    #pragma unroll
    for (int i = 0; i < 2; ++i) {
        int mb = wm * 32 + i * 16 + lg * 8;
        if (mb >= 96) continue;
        #pragma unroll
        for (int j = 0; j < 4; ++j) {
            int n = n0 + wn * 64 + j * 16 + lh;
            union { bf16 h[8]; int4 q; } u;
            #pragma unroll
            for (int v = 0; v < 8; ++v) u.h[v] = f2bf(acc[i][j][v]);
            *(int4*)(Pb + (size_t)n * 96 + mb) = u.q;
        }
    }
}

// ---------------------------------------------------------------------------
// G2: h[b][t][o] = leaky(obs@op_w + P@Bc[b] + op_b) -> bf16 hpad[b][t+1][o]
// ---------------------------------------------------------------------------
__global__ __launch_bounds__(256) void k_gemm_h(
    const bf16* __restrict__ opwT, const bf16* __restrict__ obs_bf,
    const bf16* __restrict__ BcT,  const bf16* __restrict__ P,
    const float* __restrict__ op_b, bf16* __restrict__ hpad)
{
    __shared__ __align__(16) bf16 As0[128 * LDSW];
    __shared__ __align__(16) bf16 Bs0[128 * LDSW];
    __shared__ __align__(16) bf16 As1[128 * LDSW];
    __shared__ __align__(16) bf16 Bs1[128 * LDSW];
    int b = blockIdx.z, m0 = blockIdx.x * 128, n0 = blockIdx.y * 128;
    v8f acc[2][4];
    v8f z = {0,0,0,0,0,0,0,0};
    for (int i = 0; i < 2; ++i) for (int j = 0; j < 4; ++j) acc[i][j] = z;

    gemm_acc_async(As0, Bs0, As1, Bs1, opwT, 512,
                   obs_bf + (size_t)b * 512 * 512, 512,
                   512, m0, n0, acc);
    gemm_acc_async(As0, Bs0, As1, Bs1, BcT + (size_t)b * 512 * 96, 96,
                   P + (size_t)b * 512 * 96, 96,
                   96, m0, n0, acc);

    const int wave = threadIdx.x >> 5, lane = threadIdx.x & 31;
    const int wm = wave >> 1, wn = wave & 1, lh = lane & 15, lg = lane >> 4;
    bf16* Hb = hpad + (size_t)b * 514 * 512;
    #pragma unroll
    for (int i = 0; i < 2; ++i) {
        int mb = m0 + wm * 32 + i * 16 + lg * 8;
        #pragma unroll
        for (int j = 0; j < 4; ++j) {
            int n = n0 + wn * 64 + j * 16 + lh;
            union { bf16 h[8]; int4 q; } u;
            #pragma unroll
            for (int v = 0; v < 8; ++v) {
                float x = acc[i][j][v] + op_b[mb + v];
                x = x > 0.f ? x : 0.01f * x;
                u.h[v] = f2bf(x);
            }
            *(int4*)(Hb + (size_t)(n + 1) * 512 + mb) = u.q;
        }
    }
}

// ---------------------------------------------------------------------------
// Conv1d(512->512,k=3,'same') = 3 shifted GEMMs; input bf16 [B][514][512].
// ---------------------------------------------------------------------------
__global__ __launch_bounds__(256) void k_conv(
    const bf16* __restrict__ W,      // [3][512][512]  W[k][o][i]
    const bf16* __restrict__ Xpad,   // [B][514][512]
    const float* __restrict__ bias,
    bf16* __restrict__ Y, int yRowOff, long long yBStride)
{
    __shared__ __align__(16) bf16 As0[128 * LDSW];
    __shared__ __align__(16) bf16 Bs0[128 * LDSW];
    __shared__ __align__(16) bf16 As1[128 * LDSW];
    __shared__ __align__(16) bf16 Bs1[128 * LDSW];
    int b = blockIdx.z, m0 = blockIdx.x * 128, n0 = blockIdx.y * 128;
    v8f acc[2][4];
    v8f z = {0,0,0,0,0,0,0,0};
    for (int i = 0; i < 2; ++i) for (int j = 0; j < 4; ++j) acc[i][j] = z;

    const bf16* Xb = Xpad + (size_t)b * 514 * 512;
    for (int k = 0; k < 3; ++k)
        gemm_acc_async(As0, Bs0, As1, Bs1,
                       W + (size_t)k * 512 * 512, 512,
                       Xb + (size_t)k * 512, 512,   // row n -> padded row n+k
                       512, m0, n0, acc);

    const int wave = threadIdx.x >> 5, lane = threadIdx.x & 31;
    const int wm = wave >> 1, wn = wave & 1, lh = lane & 15, lg = lane >> 4;
    bf16* Yb = Y + (size_t)b * yBStride;
    #pragma unroll
    for (int i = 0; i < 2; ++i) {
        int mb = m0 + wm * 32 + i * 16 + lg * 8;
        #pragma unroll
        for (int j = 0; j < 4; ++j) {
            int n = n0 + wn * 64 + j * 16 + lh;
            union { bf16 h[8]; int4 q; } u;
            #pragma unroll
            for (int v = 0; v < 8; ++v) {
                float x = acc[i][j][v] + bias[mb + v];
                x = x > 0.f ? x : 0.01f * x;
                u.h[v] = f2bf(x);
            }
            *(int4*)(Yb + (size_t)(n + yRowOff) * 512 + mb) = u.q;
        }
    }
}

// ---------------------------------------------------------------------------
__global__ __launch_bounds__(512) void k_max(const bf16* __restrict__ x3, float* __restrict__ m) {
    int b = blockIdx.x, o = threadIdx.x;
    const bf16* p = x3 + (size_t)b * 512 * 512 + o;
    float best = -3.0e38f;
    for (int t = 0; t < 512; ++t) best = fmaxf(best, (float)p[(size_t)t * 512]);
    m[b * 512 + o] = best;
}

// ---------------------------------------------------------------------------
__global__ __launch_bounds__(512) void k_head1(
    const float* __restrict__ m, const float* __restrict__ rw,
    const float* __restrict__ w, const float* __restrict__ bias,
    const float* __restrict__ As_, const float* __restrict__ Bs_,
    const float* __restrict__ Ae, const float* __restrict__ Be,
    float* __restrict__ x4)
{
    int b = blockIdx.x, o = threadIdx.x;
    __shared__ float mv[512];
    __shared__ float s[72];
    mv[o] = m[b * 512 + o];
    __syncthreads();
    if (o < 72) {
        float acc = 0.f;
        if (o < 8) { for (int i = 0; i < 512; ++i) acc += mv[i] * As_[i * 8 + o]; }
        else {
            int e = (o - 8) >> 3, r = (o - 8) & 7;
            const float* A = Ae + (size_t)e * 512 * 8 + r;
            for (int i = 0; i < 512; ++i) acc += mv[i] * A[(size_t)i * 8];
        }
        s[o] = acc;
    }
    __syncthreads();
    float acc = bias[o];
    for (int i = 0; i < 512; ++i) acc += mv[i] * w[i * 512 + o];
    for (int r = 0; r < 8; ++r) acc += s[r] * Bs_[r * 512 + o];
    for (int e = 0; e < 8; ++e) {
        float rwe = rw[b * 8 + e];
        for (int r = 0; r < 8; ++r) acc += rwe * s[8 + e * 8 + r] * Be[(size_t)(e * 8 + r) * 512 + o];
    }
    x4[b * 512 + o] = acc > 0.f ? acc : 0.01f * acc;
}

// ---------------------------------------------------------------------------
__global__ __launch_bounds__(128) void k_head2(
    const float* __restrict__ x4, const float* __restrict__ rw,
    const float* __restrict__ w, const float* __restrict__ bias,
    const float* __restrict__ As_, const float* __restrict__ Bs_,
    const float* __restrict__ Ae, const float* __restrict__ Be,
    float* __restrict__ out)
{
    int b = blockIdx.x, t = threadIdx.x;
    __shared__ float s[73];
    const float* xb = x4 + b * 512;
    if (t < 73) {
        float acc = 0.f;
        if (t == 72)      { for (int i = 0; i < 512; ++i) acc += xb[i] * w[i]; }
        else if (t < 8)   { for (int i = 0; i < 512; ++i) acc += xb[i] * As_[i * 8 + t]; }
        else {
            int e = (t - 8) >> 3, r = (t - 8) & 7;
            const float* A = Ae + (size_t)e * 512 * 8 + r;
            for (int i = 0; i < 512; ++i) acc += xb[i] * A[(size_t)i * 8];
        }
        s[t] = acc;
    }
    __syncthreads();
    if (t == 0) {
        float acc = s[72] + bias[0];
        for (int r = 0; r < 8; ++r) acc += s[r] * Bs_[r];
        for (int e = 0; e < 8; ++e) {
            float rwe = rw[b * 8 + e];
            for (int r = 0; r < 8; ++r) acc += rwe * s[8 + e * 8 + r] * Be[e * 8 + r];
        }
        out[b] = acc;
    }
}

// ---------------------------------------------------------------------------
static inline size_t alignUp(size_t x) { return (x + 255) & ~(size_t)255; }

extern "C" void kernel_launch(void* const* d_in, const int* in_sizes, int n_in,
                              void* d_out, int out_size, void* d_ws, size_t ws_size,
                              hipStream_t stream) {
    const float* user_emb = (const float*)d_in[0];
    const float* obs      = (const float*)d_in[1];
    const float* gw1      = (const float*)d_in[2];
    const float* gb1      = (const float*)d_in[3];
    const float* gw2      = (const float*)d_in[4];
    const float* gb2      = (const float*)d_in[5];
    const float* op_w     = (const float*)d_in[6];
    const float* op_b     = (const float*)d_in[7];
    const float* op_As    = (const float*)d_in[8];
    const float* op_Bs    = (const float*)d_in[9];
    const float* op_Ae    = (const float*)d_in[10];
    const float* op_Be    = (const float*)d_in[11];
    const float* cw1      = (const float*)d_in[12];
    const float* cb1      = (const float*)d_in[13];
    const float* cw2      = (const float*)d_in[14];
    const float* cb2      = (const float*)d_in[15];
    const float* h1_w     = (const float*)d_in[16];
    const float* h1_b     = (const float*)d_in[17];
    const float* h1_As    = (const float*)d_in[18];
    const float* h1_Bs    = (const float*)d_in[19];
    const float* h1_Ae    = (const float*)d_in[20];
    const float* h1_Be    = (const float*)d_in[21];
    const float* h2_w     = (const float*)d_in[22];
    const float* h2_b     = (const float*)d_in[23];
    const float* h2_As    = (const float*)d_in[24];
    const float* h2_Bs    = (const float*)d_in[25];
    const float* h2_Ae    = (const float*)d_in[26];
    const float* h2_Be    = (const float*)d_in[27];
    float* out = (float*)d_out;

    // ---- workspace carve (all 256B aligned) ----
    char* p = (char*)d_ws;
    size_t off = 0;
    auto carve = [&](size_t bytes) { void* r = p + off; off = alignUp(off + bytes); return r; };
    float* rw    = (float*)carve(64 * 8 * sizeof(float));
    bf16* obs_bf = (bf16*)carve((size_t)64 * 512 * 512 * 2);
    bf16* opwT   = (bf16*)carve((size_t)512 * 512 * 2);
    bf16* AallT  = (bf16*)carve((size_t)96 * 512 * 2);
    bf16* BcT    = (bf16*)carve((size_t)64 * 512 * 96 * 2);
    bf16* P      = (bf16*)carve((size_t)64 * 512 * 96 * 2);
    bf16* hpad   = (bf16*)carve((size_t)64 * 514 * 512 * 2);
    bf16* w1k    = (bf16*)carve((size_t)3 * 512 * 512 * 2);
    bf16* w2k    = (bf16*)carve((size_t)3 * 512 * 512 * 2);
    bf16* h2pad  = (bf16*)carve((size_t)64 * 514 * 512 * 2);
    bf16* x3     = (bf16*)carve((size_t)64 * 512 * 512 * 2);
    float* mpool = (float*)carve((size_t)64 * 512 * sizeof(float));
    float* x4    = (float*)carve((size_t)64 * 512 * sizeof(float));
    (void)ws_size; (void)n_in; (void)in_sizes; (void)out_size;

    // ---- pipeline ----
    k_gating<<<64, 128, 0, stream>>>(user_emb, gw1, gb1, gw2, gb2, rw);

    k_cvt_bf16<<<8192, 256, 0, stream>>>(obs, obs_bf, (long long)64 * 512 * 512);
    k_pack_opwT<<<1024, 256, 0, stream>>>(op_w, opwT);
    k_pack_AallT<<<192, 256, 0, stream>>>(op_As, op_Ae, AallT);
    k_pack_BcT<<<12288, 256, 0, stream>>>(op_Bs, op_Be, rw, BcT);
    k_pack_convw<<<3072, 256, 0, stream>>>(cw1, w1k);
    k_pack_convw<<<3072, 256, 0, stream>>>(cw2, w2k);
    k_zero_pads<<<256, 256, 0, stream>>>(hpad, h2pad);

    k_gemm_p<<<dim3(4, 64), 256, 0, stream>>>(AallT, obs_bf, P);
    k_gemm_h<<<dim3(4, 4, 64), 256, 0, stream>>>(opwT, obs_bf, BcT, P, op_b, hpad);

    k_conv<<<dim3(4, 4, 64), 256, 0, stream>>>(w1k, hpad,  cb1, h2pad, 1, (long long)514 * 512);
    k_conv<<<dim3(4, 4, 64), 256, 0, stream>>>(w2k, h2pad, cb2, x3,    0, (long long)512 * 512);

    k_max<<<64, 512, 0, stream>>>(x3, mpool);
    k_head1<<<64, 512, 0, stream>>>(mpool, rw, h1_w, h1_b, h1_As, h1_Bs, h1_Ae, h1_Be, x4);
    k_head2<<<64, 128, 0, stream>>>(x4, rw, h2_w, h2_b, h2_As, h2_Bs, h2_Ae, h2_Be, out);
}
